// M3GNet_47132971106999
// MI455X (gfx1250) — compile-verified, matrix operand order checked by reading, not run on hardware
//
#include <hip/hip_runtime.h>
#include <hip/hip_bf16.h>
#include <math.h>

typedef _Float16 half_t;
typedef __attribute__((ext_vector_type(16))) _Float16 v16h;
typedef __attribute__((ext_vector_type(8)))  _Float16 v8h;
typedef __attribute__((ext_vector_type(4)))  _Float16 v4h;
typedef __attribute__((ext_vector_type(8)))  float    v8f;
typedef __attribute__((ext_vector_type(4)))  float    v4f;

__device__ __forceinline__ float sigmf(float x){
  // fast sigmoid: v_exp_f32 + v_rcp_f32 (avoid IEEE div expansion)
  return __builtin_amdgcn_rcpf(1.0f + __expf(-x));
}
__device__ __forceinline__ float siluf(float x){ return x * sigmf(x); }

__device__ __forceinline__ v4h cvt4(v4f x){
  v4h h;
#pragma unroll
  for (int i = 0; i < 4; ++i) h[i] = (half_t)x[i];
  return h;
}

// Load a 16-half WMMA fragment (A or B) from a row pointer.
// 16-bit A 16x32 layout (ISA 7.12.2): lanes 0-15 hold K = [koff..koff+7] and
// [koff+16..koff+23] with koff = 32*step + 8*(lane>=16). Two 16B contiguous reads.
__device__ __forceinline__ v16h load_frag(const half_t* row, int koff){
  v8h lo = *(const v8h*)(row + koff);
  v8h h8 = *(const v8h*)(row + koff + 16);
  v16h f;
#pragma unroll
  for (int i = 0; i < 8; ++i){ f[i] = lo[i]; f[8+i] = h8[i]; }
  return f;
}

// ---------------- init kernels ----------------

__global__ void k_atom_init(const int* __restrict__ types, const float* __restrict__ emb,
                            float* __restrict__ atom, int n16){
  for (int idx4 = blockIdx.x*blockDim.x + threadIdx.x; idx4 < n16; idx4 += gridDim.x*blockDim.x){
    int n = idx4 >> 4, c4 = idx4 & 15;
    *(v4f*)(atom + (size_t)idx4*4) = *(const v4f*)(emb + (size_t)types[n]*64 + c4*4);
  }
}

__global__ void k_bond_init(const float* __restrict__ bf, const float* __restrict__ Wb,
                            const float* __restrict__ bb, float* __restrict__ bond, int E){
  __shared__ float sW[192];
  __shared__ float sb[64];
  int tid = threadIdx.x;
  if (tid < 192) sW[tid] = Wb[tid];
  if (tid < 64)  sb[tid] = bb[tid];
  __syncthreads();
  int total4 = E*16;
  for (int idx4 = blockIdx.x*blockDim.x + tid; idx4 < total4; idx4 += gridDim.x*blockDim.x){
    int e = idx4 >> 4, c4 = (idx4 & 15)*4;
    float x0 = bf[e*3+0], x1 = bf[e*3+1], x2 = bf[e*3+2];
    v4f o;
#pragma unroll
    for (int i = 0; i < 4; ++i){
      int c = c4 + i;
      o[i] = siluf(x0*sW[c] + x1*sW[64+c] + x2*sW[128+c] + sb[c]);
    }
    *(v4f*)(bond + (size_t)idx4*4) = o;
  }
}

// transpose (KxC f32) -> (CxKpad f16), zero-padding K
__global__ void k_tpad(const float* __restrict__ src, half_t* __restrict__ dst,
                       int Kin, int Kpad, int C){
  int idx = blockIdx.x*blockDim.x + threadIdx.x;
  int total = C*Kpad;
  if (idx >= total) return;
  int c = idx / Kpad, k = idx - c*Kpad;
  dst[idx] = (half_t)((k < Kin) ? src[k*C + c] : 0.0f);
}

// ---------------- three-body scatter (basis recomputed on the fly) ----------------
// 32 triples staged per wave with coalesced index/scalar loads, then each triple
// broadcast via shuffles; 32 lanes cover the 48 basis elements.
__global__ void k_contrib(const int* __restrict__ tbi, const int* __restrict__ bidx,
                          const float* __restrict__ tlen, const float* __restrict__ tfeat,
                          const float* __restrict__ w, float* __restrict__ agg, int T){
  const float PI = 3.14159265358979f;
  int lane = threadIdx.x & 31;
  int wid = (blockIdx.x*blockDim.x + threadIdx.x) >> 5;
  int nw  = (gridDim.x*blockDim.x) >> 5;
  for (int base = wid*32; base < T; base += nw*32){
    int t = base + lane;
    int tb = 0, ak = 0;
    float r = 1.0f, c = 0.0f;
    if (t < T){
      tb = tbi[2*t];
      int tb1 = tbi[2*t + 1];
      ak = bidx[2*tb1 + 1];            // atom_k = bj[tbi[:,1]]
      r = tlen[t];
      c = tfeat[t];
    }
    int cnt = T - base; if (cnt > 32) cnt = 32;
    for (int i = 0; i < cnt; ++i){
      int   tb_i = __shfl(tb, i, 32);
      int   ak_i = __shfl(ak, i, 32);
      float r_i  = __shfl(r,  i, 32);
      float c_i  = __shfl(c,  i, 32);
      float u = r_i * 0.25f;           // r / RC3
      float u2 = u*u, u3 = u2*u;
      float fc = 1.0f - 6.0f*u3*u2 + 15.0f*u2*u2 - 10.0f*u3;
      float rinv = fc / fmaxf(r_i, 1e-6f);
      float P1 = c_i;
      float P2 = 0.5f*(3.0f*c_i*c_i - 1.0f);
      float P3 = 0.5f*(5.0f*c_i*c_i*c_i - 3.0f*c_i);
      const float* wrow = w + (size_t)ak_i*48;
      float* arow = agg + (size_t)tb_i*48;
      for (int e = lane; e < 48; e += 32){
        int nI = e >> 4, l = e & 15;
        float rad = __sinf((float)(nI+1) * PI * u) * rinv;
        float ang = (l == 0) ? 1.0f : (l < 4) ? P1 : (l < 9) ? P2 : P3;
        atomicAdd(arow + e, rad * ang * wrow[e]);
      }
    }
  }
}

// ---------------- WMMA kernels ----------------

// w = sigmoid(atom @ W_up + b_up): K=64, 48 cols -> 3 col tiles, 4 waves/block
__global__ void k_wup(const float* __restrict__ atom, const half_t* __restrict__ wt,
                      const float* __restrict__ bvec, float* __restrict__ w, int ntiles){
  __shared__ __align__(16) half_t ldsA[16*64];
  __shared__ float ldsO[3][16][16];
  const int tid = threadIdx.x, wave = tid >> 5, lane = tid & 31;
  const int n = lane & 15, hi = lane >> 4;
  v16h b0 = {}, b1 = {};
  float bias = 0.0f;
  if (wave < 3){
    const half_t* brow = wt + (wave*16 + n)*64;
    b0 = load_frag(brow, hi*8);
    b1 = load_frag(brow, 32 + hi*8);
    bias = bvec[wave*16 + n];
  }
  const half_t* arow = ldsA + n*64;
  const int orow = tid & 15, ogrp = tid >> 4;   // 16 rows x 8 groups of 6 cols
  for (int tile = blockIdx.x; tile < ntiles; tile += gridDim.x){
    const int rowbase = tile*16;
#pragma unroll
    for (int j = 0; j < 2; ++j){
      int idx4 = tid + 128*j;          // 256 float4 = 16 rows x 64 floats
      v4f x = *(const v4f*)(atom + (size_t)rowbase*64 + (size_t)idx4*4);
      *(v4h*)(ldsA + idx4*4) = cvt4(x);
    }
    __syncthreads();
    if (wave < 3){
      v8f acc;
#pragma unroll
      for (int i = 0; i < 8; ++i) acc[i] = bias;
      v16h a0 = load_frag(arow, hi*8);
      v16h a1 = load_frag(arow, 32 + hi*8);
      acc = __builtin_amdgcn_wmma_f32_16x16x32_f16(false, a0, false, b0, (short)0, acc, false, false);
      acc = __builtin_amdgcn_wmma_f32_16x16x32_f16(false, a1, false, b1, (short)0, acc, false, false);
#pragma unroll
      for (int v = 0; v < 8; ++v) ldsO[wave][v + 8*hi][n] = acc[v];
    }
    __syncthreads();
    {
      float* wrow = w + (size_t)(rowbase + orow)*48 + ogrp*6;
#pragma unroll
      for (int j = 0; j < 6; ++j){
        int cc = ogrp*6 + j;
        wrow[j] = sigmf(ldsO[cc >> 4][orow][cc & 15]);
      }
    }
    __syncthreads();
  }
}

// bond += silu(agg@W3g+b3g)*sigmoid(agg@W3s+b3s): K=48 padded to 64
__global__ void k_gated48(const float* __restrict__ agg, float* __restrict__ bond,
                          const half_t* __restrict__ wtg, const half_t* __restrict__ wts,
                          const float* __restrict__ bg, const float* __restrict__ bs, int ntiles){
  __shared__ __align__(16) half_t ldsA[16*64];
  __shared__ float ldsO[8][16][16];
  const int tid = threadIdx.x, wave = tid >> 5, lane = tid & 31;
  const int n = lane & 15, hi = lane >> 4;
  const half_t* wt = (wave < 4) ? wtg : wts;
  const float* bv = (wave < 4) ? bg : bs;
  const int ct = wave & 3;
  const half_t* brow = wt + (ct*16 + n)*64;
  v16h b0 = load_frag(brow, hi*8);
  v16h b1 = load_frag(brow, 32 + hi*8);
  const float bias = bv[ct*16 + n];
  const half_t* arow = ldsA + n*64;
  const int r = tid >> 4, q = tid & 15;          // (row, float4-slot)
  const int t0 = q >> 2, c0 = (q & 3)*4;         // ldsO tile + col base for this quad
  for (int tile = blockIdx.x; tile < ntiles; tile += gridDim.x){
    const int rowbase = tile*16;
    {   // prefetch next grid-strided tile of agg (16*48*4 = 3 KB = 24 x 128B lines)
      int nt = tile + gridDim.x;
      if (nt < ntiles && tid < 24)
        __builtin_prefetch((const char*)(agg + (size_t)nt*16*48) + tid*128, 0, 1);
    }
    {   // 16 rows x 16 float4 (last 4 float4 of each row are zero pad)
      v4h h = {};
      if (q < 12)
        h = cvt4(*(const v4f*)(agg + (size_t)(rowbase + r)*48 + q*4));
      *(v4h*)(ldsA + (r*64 + q*4)) = h;
    }
    __syncthreads();
    v8f acc;
#pragma unroll
    for (int i = 0; i < 8; ++i) acc[i] = bias;
    v16h a0 = load_frag(arow, hi*8);
    v16h a1 = load_frag(arow, 32 + hi*8);
    acc = __builtin_amdgcn_wmma_f32_16x16x32_f16(false, a0, false, b0, (short)0, acc, false, false);
    acc = __builtin_amdgcn_wmma_f32_16x16x32_f16(false, a1, false, b1, (short)0, acc, false, false);
#pragma unroll
    for (int v = 0; v < 8; ++v) ldsO[wave][v + 8*hi][n] = acc[v];
    __syncthreads();
    {   // vector read-modify-write of one float4 of bond per thread
      float* bq = bond + (size_t)(rowbase + r)*64 + q*4;
      v4f x = *(const v4f*)bq;
#pragma unroll
      for (int i = 0; i < 4; ++i){
        float g = ldsO[t0][r][c0 + i];
        float s = ldsO[4 + t0][r][c0 + i];
        x[i] += siluf(g) * sigmf(s);
      }
      *(v4f*)bq = x;
    }
    __syncthreads();
  }
}

// bond += gated(concat(atom[bi],atom[bj],bond) @ Wb{g,s}): K=192, gathers fused into A stage
__global__ void k_gated_feat(const float* __restrict__ atom, const int* __restrict__ bidx,
                             float* __restrict__ bond,
                             const half_t* __restrict__ wtg, const half_t* __restrict__ wts,
                             const float* __restrict__ bg, const float* __restrict__ bs, int ntiles){
  __shared__ __align__(16) half_t ldsA[16*192];
  __shared__ float ldsO[8][16][16];
  const int tid = threadIdx.x, wave = tid >> 5, lane = tid & 31;
  const int n = lane & 15, hi = lane >> 4;
  const half_t* wt = (wave < 4) ? wtg : wts;
  const float* bv = (wave < 4) ? bg : bs;
  const int ct = wave & 3;
  const half_t* brow = wt + (ct*16 + n)*192;
  v16h bfr[6];
#pragma unroll
  for (int s = 0; s < 6; ++s) bfr[s] = load_frag(brow, 32*s + hi*8);
  const float bias = bv[ct*16 + n];
  const half_t* arow = ldsA + n*192;
  const int r = tid >> 4, q = tid & 15;
  const int t0 = q >> 2, c0 = (q & 3)*4;
  for (int tile = blockIdx.x; tile < ntiles; tile += gridDim.x){
    const int rowbase = tile*16;
    {   // prefetch next tile's bond rows (4 KB) and edge indices
      int nt = tile + gridDim.x;
      if (nt < ntiles){
        if (tid < 32)
          __builtin_prefetch((const char*)(bond + (size_t)nt*16*64) + tid*128, 0, 1);
        else if (tid == 32)
          __builtin_prefetch(bidx + (size_t)2*nt*16, 0, 1);
      }
    }
    {   // each thread: one row r, float4 slot q, all three 64-wide regions
      const int grow = rowbase + r;
      const int ai = bidx[2*grow], aj = bidx[2*grow + 1];
      v4f x0 = *(const v4f*)(atom + (size_t)ai*64 + q*4);
      v4f x1 = *(const v4f*)(atom + (size_t)aj*64 + q*4);
      v4f x2 = *(const v4f*)(bond + (size_t)grow*64 + q*4);
      half_t* dst = ldsA + r*192 + q*4;
      *(v4h*)(dst)        = cvt4(x0);
      *(v4h*)(dst + 64)   = cvt4(x1);
      *(v4h*)(dst + 128)  = cvt4(x2);
    }
    __syncthreads();
    v8f acc;
#pragma unroll
    for (int i = 0; i < 8; ++i) acc[i] = bias;
#pragma unroll
    for (int s = 0; s < 6; ++s){
      v16h a = load_frag(arow, 32*s + hi*8);
      acc = __builtin_amdgcn_wmma_f32_16x16x32_f16(false, a, false, bfr[s], (short)0, acc, false, false);
    }
#pragma unroll
    for (int v = 0; v < 8; ++v) ldsO[wave][v + 8*hi][n] = acc[v];
    __syncthreads();
    {
      float* bq = bond + (size_t)(rowbase + r)*64 + q*4;
      v4f x = *(const v4f*)bq;
#pragma unroll
      for (int i = 0; i < 4; ++i){
        float g = ldsO[t0][r][c0 + i];
        float s = ldsO[4 + t0][r][c0 + i];
        x[i] += siluf(g) * sigmf(s);
      }
      *(v4f*)bq = x;
    }
    __syncthreads();
  }
}

// atom[bi] += gated(bond @ Wa{g,s}): K=64, scatter-add output
__global__ void k_gated_msg(const float* __restrict__ bond, const int* __restrict__ bidx,
                            float* __restrict__ atom,
                            const half_t* __restrict__ wtg, const half_t* __restrict__ wts,
                            const float* __restrict__ bg, const float* __restrict__ bs, int ntiles){
  __shared__ __align__(16) half_t ldsA[16*64];
  __shared__ float ldsO[8][16][16];
  const int tid = threadIdx.x, wave = tid >> 5, lane = tid & 31;
  const int n = lane & 15, hi = lane >> 4;
  const half_t* wt = (wave < 4) ? wtg : wts;
  const float* bv = (wave < 4) ? bg : bs;
  const int ct = wave & 3;
  const half_t* brow = wt + (ct*16 + n)*64;
  v16h b0 = load_frag(brow, hi*8);
  v16h b1 = load_frag(brow, 32 + hi*8);
  const float bias = bv[ct*16 + n];
  const half_t* arow = ldsA + n*64;
  const int r = tid >> 4, q = tid & 15;
  const int t0 = q >> 2, c0 = (q & 3)*4;
  for (int tile = blockIdx.x; tile < ntiles; tile += gridDim.x){
    const int rowbase = tile*16;
    {   // prefetch next tile's bond rows (4 KB)
      int nt = tile + gridDim.x;
      if (nt < ntiles && tid < 32)
        __builtin_prefetch((const char*)(bond + (size_t)nt*16*64) + tid*128, 0, 1);
    }
    {   // 256 float4 = full 16x64 tile, one per thread
      v4f x = *(const v4f*)(bond + (size_t)rowbase*64 + (size_t)tid*4);
      *(v4h*)(ldsA + tid*4) = cvt4(x);
    }
    __syncthreads();
    v8f acc;
#pragma unroll
    for (int i = 0; i < 8; ++i) acc[i] = bias;
    v16h a0 = load_frag(arow, hi*8);
    v16h a1 = load_frag(arow, 32 + hi*8);
    acc = __builtin_amdgcn_wmma_f32_16x16x32_f16(false, a0, false, b0, (short)0, acc, false, false);
    acc = __builtin_amdgcn_wmma_f32_16x16x32_f16(false, a1, false, b1, (short)0, acc, false, false);
#pragma unroll
    for (int v = 0; v < 8; ++v) ldsO[wave][v + 8*hi][n] = acc[v];
    __syncthreads();
    {   // one bi lookup per thread, 4 unrolled f32 atomics into the target atom row
      float* arow_out = atom + (size_t)bidx[2*(rowbase + r)]*64 + q*4;
#pragma unroll
      for (int i = 0; i < 4; ++i){
        float g = ldsO[t0][r][c0 + i];
        float s = ldsO[4 + t0][r][c0 + i];
        atomicAdd(arow_out + i, siluf(g) * sigmf(s));
      }
    }
    __syncthreads();
  }
}

// ---------------- readout ----------------

__global__ void k_prefix(const int* __restrict__ n_atoms, int* __restrict__ off, int G){
  if (blockIdx.x == 0 && threadIdx.x == 0){
    int s = 0; off[0] = 0;
    for (int g = 0; g < G; ++g){ s += n_atoms[g]; off[g+1] = s; }
  }
}

__global__ void k_readout(const float* __restrict__ atom, const float* __restrict__ Wr,
                          const float* __restrict__ br, const int* __restrict__ off,
                          float* __restrict__ out){
  __shared__ float sW[64];
  __shared__ float red[256];
  int g = blockIdx.x;
  if (threadIdx.x < 64) sW[threadIdx.x] = Wr[threadIdx.x];
  __syncthreads();
  int a0 = off[g], a1 = off[g+1];
  float acc = 0.0f;
  for (int a = a0 + threadIdx.x; a < a1; a += blockDim.x){
    const float* ar = atom + (size_t)a*64;
    float d = 0.0f;
#pragma unroll
    for (int k = 0; k < 64; ++k) d += ar[k]*sW[k];
    acc += d;
  }
  red[threadIdx.x] = acc;
  __syncthreads();
  for (int s = 128; s > 0; s >>= 1){
    if (threadIdx.x < s) red[threadIdx.x] += red[threadIdx.x + s];
    __syncthreads();
  }
  if (threadIdx.x == 0) out[g] = red[0] + (float)(a1 - a0)*br[0];
}

// ---------------- host ----------------

extern "C" void kernel_launch(void* const* d_in, const int* in_sizes, int n_in,
                              void* d_out, int out_size, void* d_ws, size_t ws_size,
                              hipStream_t stream){
  const int*   atom_types    = (const int*)d_in[0];
  const float* bond_features = (const float*)d_in[1];
  const int*   bond_indices  = (const int*)d_in[2];
  const int*   tbi           = (const int*)d_in[3];
  const float* tlen          = (const float*)d_in[4];
  const float* tfeat         = (const float*)d_in[5];
  const int*   n_atoms       = (const int*)d_in[6];
  const float* emb           = (const float*)d_in[7];
  const float* Wb            = (const float*)d_in[8];
  const float* bb            = (const float*)d_in[9];
  const float* W_up          = (const float*)d_in[10];
  const float* b_up          = (const float*)d_in[11];
  const float* W3g           = (const float*)d_in[12];
  const float* b3g           = (const float*)d_in[13];
  const float* W3s           = (const float*)d_in[14];
  const float* b3s           = (const float*)d_in[15];
  const float* Wbg           = (const float*)d_in[16];
  const float* bbg           = (const float*)d_in[17];
  const float* Wbs           = (const float*)d_in[18];
  const float* bbs           = (const float*)d_in[19];
  const float* Wag           = (const float*)d_in[20];
  const float* bag           = (const float*)d_in[21];
  const float* Was           = (const float*)d_in[22];
  const float* bas           = (const float*)d_in[23];
  const float* Wr            = (const float*)d_in[24];
  const float* br            = (const float*)d_in[25];
  float* out = (float*)d_out;
  (void)n_in; (void)out_size; (void)ws_size;

  const int N = in_sizes[0];
  const int E = in_sizes[2] / 2;
  const int T = in_sizes[4];
  const int G = in_sizes[6];

  char* base = (char*)d_ws;
  size_t woff = 0;
  auto carve = [&](size_t bytes) -> char* {
    char* p = base + woff;
    woff = (woff + bytes + 255) & ~(size_t)255;
    return p;
  };
  float* atom = (float*)carve((size_t)N*64*4);
  float* bond = (float*)carve((size_t)E*64*4);
  float* wbuf = (float*)carve((size_t)N*48*4);
  float* agg  = (float*)carve((size_t)E*48*4);
  int*   goff = (int*)carve((size_t)(G+1)*4);
  half_t *wt_up[3], *wt3g[3], *wt3s[3], *wtbg[3], *wtbs[3], *wtag[3], *wtas[3];
  for (int b = 0; b < 3; ++b){
    wt_up[b] = (half_t*)carve(48*64*2);
    wt3g[b]  = (half_t*)carve(64*64*2);
    wt3s[b]  = (half_t*)carve(64*64*2);
    wtbg[b]  = (half_t*)carve(64*192*2);
    wtbs[b]  = (half_t*)carve(64*192*2);
    wtag[b]  = (half_t*)carve(64*64*2);
    wtas[b]  = (half_t*)carve(64*64*2);
  }

  k_atom_init<<<512, 256, 0, stream>>>(atom_types, emb, atom, N*16);
  k_bond_init<<<2048, 256, 0, stream>>>(bond_features, Wb, bb, bond, E);

  auto tpad = [&](const float* src, half_t* dst, int Kin, int Kpad, int C){
    int total = C*Kpad;
    k_tpad<<<(total + 255)/256, 256, 0, stream>>>(src, dst, Kin, Kpad, C);
  };
  for (int b = 0; b < 3; ++b){
    tpad(W_up + b*64*48,  wt_up[b], 64, 64, 48);
    tpad(W3g  + b*48*64,  wt3g[b],  48, 64, 64);
    tpad(W3s  + b*48*64,  wt3s[b],  48, 64, 64);
    tpad(Wbg  + b*192*64, wtbg[b], 192, 192, 64);
    tpad(Wbs  + b*192*64, wtbs[b], 192, 192, 64);
    tpad(Wag  + b*64*64,  wtag[b],  64, 64, 64);
    tpad(Was  + b*64*64,  wtas[b],  64, 64, 64);
  }

  const int ntilesN = N/16, ntilesE = E/16;
  for (int b = 0; b < 3; ++b){
    k_wup<<<1024, 128, 0, stream>>>(atom, wt_up[b], b_up + b*48, wbuf, ntilesN);
    hipMemsetAsync(agg, 0, (size_t)E*48*4, stream);
    k_contrib<<<4096, 256, 0, stream>>>(tbi, bond_indices, tlen, tfeat, wbuf, agg, T);
    k_gated48<<<2048, 256, 0, stream>>>(agg, bond, wt3g[b], wt3s[b], b3g + b*64, b3s + b*64, ntilesE);
    k_gated_feat<<<2048, 256, 0, stream>>>(atom, bond_indices, bond, wtbg[b], wtbs[b],
                                           bbg + b*64, bbs + b*64, ntilesE);
    k_gated_msg<<<2048, 256, 0, stream>>>(bond, bond_indices, atom, wtag[b], wtas[b],
                                          bag + b*64, bas + b*64, ntilesE);
  }
  k_prefix<<<1, 32, 0, stream>>>(n_atoms, goff, G);
  k_readout<<<G, 256, 0, stream>>>(atom, Wr, br, goff, out);
}
